// Multi_Line_Model_on_going_78297253806565
// MI455X (gfx1250) — matrix operand order repaired and checked by reading, not compiled, not run on hardware
//
#include <hip/hip_runtime.h>
#include <hip/hip_bf16.h>

typedef __bf16 bf16_t;
typedef __attribute__((ext_vector_type(16))) __bf16 v16bf;
typedef __attribute__((ext_vector_type(8)))  __bf16 v8bf;
typedef __attribute__((ext_vector_type(8)))  float  v8f;

#define BATCH 2048
#define FEAT  512
#define HID   512
#define HW    64
#define NLINES 32
#define OUT_SZ 42
#define KGATE 576
#define NGATE 2048
#define SS_PROB 0.1f
#define BETA 2.0f

#if __has_builtin(__builtin_amdgcn_global_load_async_to_lds_b128)
#define HAVE_ASYNC_LDS 1
typedef int v4i_t __attribute__((vector_size(16)));
typedef __attribute__((address_space(1))) v4i_t* gv4i_p;
typedef __attribute__((address_space(3))) v4i_t* lv4i_p;
#define GLD_ASYNC16(dst, src)                                                  \
  __builtin_amdgcn_global_load_async_to_lds_b128((gv4i_p)(src), (lv4i_p)(dst), \
                                                 0, 0)
#if __has_builtin(__builtin_amdgcn_s_wait_asynccnt)
#define WAIT_ASYNC() __builtin_amdgcn_s_wait_asynccnt(0)
#else
#define WAIT_ASYNC() asm volatile("s_wait_asynccnt 0x0" ::: "memory")
#endif
#else
#define HAVE_ASYNC_LDS 0
#endif

// att = softmax_s(img_logit[b,:]) * BETA is step-invariant (softmax shift
// invariance kills the per-batch-scalar hid_logit and att_b), so att_feat
// is computed exactly once.
__global__ __launch_bounds__(256) void k_attfeat(const float* __restrict__ feat,
                                                 const float* __restrict__ att_w,
                                                 bf16_t* __restrict__ att_feat_bf) {
  __shared__ float part[4][HW];
  __shared__ float att[HW];
  const int b = blockIdx.x;
  const int t = threadIdx.x;
  const int s = t & 63, g = t >> 6;
  const float* fb = feat + (size_t)b * FEAT * HW;
  float acc = 0.f;
  for (int c = g; c < FEAT; c += 4) acc += fb[c * HW + s] * att_w[c];
  part[g][s] = acc;
  __syncthreads();
  if (t < HW) att[t] = part[0][t] + part[1][t] + part[2][t] + part[3][t];
  __syncthreads();
  float myv = 0.f;
  if (t < HW) {
    float m = -1e30f;
    for (int i = 0; i < HW; ++i) m = fmaxf(m, att[i]);
    float sum = 0.f;
    for (int i = 0; i < HW; ++i) sum += __expf(att[i] - m);
    myv = __expf(att[t] - m) / sum * BETA;
  }
  __syncthreads();
  if (t < HW) att[t] = myv;
  __syncthreads();
  for (int c = t; c < FEAT; c += 256) {
    float a = 0.f;
    const float* fr = fb + c * HW;
    for (int s2 = 0; s2 < HW; ++s2) a += fr[s2] * att[s2];
    att_feat_bf[(size_t)b * FEAT + c] = (bf16_t)a;
  }
}

__global__ __launch_bounds__(256) void k_prep_w(const float* __restrict__ w_ih,
                                                const float* __restrict__ w_hh,
                                                const float* __restrict__ b_ih,
                                                const float* __restrict__ b_hh,
                                                const float* __restrict__ l1w,
                                                const float* __restrict__ l2w,
                                                bf16_t* __restrict__ Wfeat,
                                                float*  __restrict__ bias_big,
                                                bf16_t* __restrict__ Wgates,
                                                bf16_t* __restrict__ lin1bf,
                                                bf16_t* __restrict__ lin2bf) {
  const size_t S0 = (size_t)NGATE * FEAT;
  const size_t S1 = NGATE;
  const size_t S2 = (size_t)NGATE * KGATE;
  const size_t S3 = (size_t)512 * 512;
  const size_t S4 = (size_t)64 * 512;
  const size_t total = S0 + S1 + S2 + S3 + S4;
  for (size_t i = (size_t)blockIdx.x * blockDim.x + threadIdx.x; i < total;
       i += (size_t)gridDim.x * blockDim.x) {
    size_t idx = i;
    if (idx < S0) {
      int n = (int)(idx / FEAT), k = (int)(idx % FEAT);
      Wfeat[idx] = (bf16_t)((n < 1536) ? w_ih[(size_t)n * 554 + k] : 0.f);
      continue;
    }
    idx -= S0;
    if (idx < S1) {
      int n = (int)idx;
      float v;
      if (n < 1024)      v = b_ih[n] + b_hh[n];
      else if (n < 1536) v = b_ih[n];
      else               v = b_hh[n - 512];
      bias_big[n] = v;
      continue;
    }
    idx -= S1;
    if (idx < S2) {
      int n = (int)(idx / KGATE), k = (int)(idx % KGATE);
      float v = 0.f;
      if (n < 1024) {
        if (k < 512)      v = w_hh[(size_t)n * 512 + k];
        else if (k < 554) v = w_ih[(size_t)n * 554 + k];
      } else if (n < 1536) {
        if (k >= 512 && k < 554) v = w_ih[(size_t)n * 554 + k];
      } else {
        if (k < 512) v = w_hh[(size_t)(n - 512) * 512 + k];
      }
      Wgates[idx] = (bf16_t)v;
      continue;
    }
    idx -= S2;
    if (idx < S3) { lin1bf[idx] = (bf16_t)l1w[idx]; continue; }
    idx -= S3;
    {
      int n = (int)(idx / 512), k = (int)(idx % 512);
      lin2bf[idx] = (bf16_t)((n < OUT_SZ) ? l2w[(size_t)n * 512 + k] : 0.f);
    }
  }
}

__global__ __launch_bounds__(256) void k_init(float* __restrict__ h,
                                              float* __restrict__ o_prev,
                                              bf16_t* __restrict__ xs_bf) {
  const size_t S0 = (size_t)BATCH * HID;
  const size_t S1 = (size_t)BATCH * OUT_SZ;
  const size_t S2 = (size_t)BATCH * KGATE;
  const size_t total = S0 + S1 + S2;
  for (size_t i = (size_t)blockIdx.x * blockDim.x + threadIdx.x; i < total;
       i += (size_t)gridDim.x * blockDim.x) {
    if (i < S0) h[i] = 0.f;
    else if (i < S0 + S1) o_prev[i - S0] = 0.f;
    else xs_bf[i - S0 - S1] = (bf16_t)0.f;
  }
}

__global__ __launch_bounds__(256) void k_ss(int t, const float* __restrict__ sp,
                                            const float* __restrict__ gt,
                                            const int* __restrict__ lp,
                                            const float* __restrict__ o_prev,
                                            bf16_t* __restrict__ xs_bf) {
  size_t i = (size_t)blockIdx.x * blockDim.x + threadIdx.x;
  if (i >= (size_t)BATCH * OUT_SZ) return;
  int b = (int)(i / OUT_SZ), j = (int)(i % OUT_SZ);
  bool up = (t == 0) || (sp[(size_t)t * BATCH + b] < SS_PROB);
  float v;
  if (up) v = o_prev[(size_t)b * OUT_SZ + j];
  else if (j < 40) v = gt[((size_t)b * NLINES + (t - 1)) * 40 + j];
  else {
    float p = (float)lp[(size_t)b * NLINES + (t - 1)];
    v = (j == 40) ? (1.f - p) : p;
  }
  xs_bf[(size_t)b * KGATE + 512 + j] = (bf16_t)v;
}

// ---------------------------------------------------------------------------
// bf16 WMMA GEMM, 128x64 block tile, 8 waves; each wave owns one 16-row strip
// and all four 16-col tiles => 4 v_wmma per A-fragment (register A reuse).
// Y[m,n] = sum_k A[m,k]*W[n,k] (+addend[m,n]) (+bias[n]); optional ReLU + bf16.
// M % 128 == 0, Npad % 64 == 0, K % 32 == 0.
// ---------------------------------------------------------------------------
__global__ __launch_bounds__(256) void k_gemm(const bf16_t* __restrict__ A,
                                              const bf16_t* __restrict__ W,
                                              const float* __restrict__ addend,
                                              const float* __restrict__ bias,
                                              float* __restrict__ Y,
                                              bf16_t* __restrict__ Ybf,
                                              int K, int Npad, int Nreal, int act) {
  __shared__ bf16_t lA[128][40];
  __shared__ bf16_t lW[64][40];
  const int t = threadIdx.x;
  const int wave = t >> 5, lane = t & 31;
  const int m0 = blockIdx.x * 128, n0 = blockIdx.y * 64;
  const int g = lane >> 4;
  const int l15 = lane & 15;
  // staging coords: A tile 128x32 = 512 x 16B (2 per thread), W tile 64x32 = 256 x 16B
  const int ar0 = t >> 1;                 // rows 0..127
  const int ac0 = (t & 1) * 16;           // col 0 or 16
  const int wr = t >> 2;                  // rows 0..63
  const int wc = (t & 3) * 8;
  v8f acc[4];
#pragma unroll
  for (int ni = 0; ni < 4; ++ni) acc[ni] = (v8f){};

  for (int kc = 0; kc < K; kc += 32) {
#if HAVE_ASYNC_LDS
    GLD_ASYNC16(&lA[ar0][ac0], &A[(size_t)(m0 + ar0) * K + kc + ac0]);
    GLD_ASYNC16(&lA[ar0][ac0 + 8], &A[(size_t)(m0 + ar0) * K + kc + ac0 + 8]);
    GLD_ASYNC16(&lW[wr][wc], &W[(size_t)(n0 + wr) * K + kc + wc]);
    WAIT_ASYNC();
#else
    *reinterpret_cast<v8bf*>(&lA[ar0][ac0]) =
        *reinterpret_cast<const v8bf*>(&A[(size_t)(m0 + ar0) * K + kc + ac0]);
    *reinterpret_cast<v8bf*>(&lA[ar0][ac0 + 8]) =
        *reinterpret_cast<const v8bf*>(&A[(size_t)(m0 + ar0) * K + kc + ac0 + 8]);
    *reinterpret_cast<v8bf*>(&lW[wr][wc]) =
        *reinterpret_cast<const v8bf*>(&W[(size_t)(n0 + wr) * K + kc + wc]);
#endif
    __syncthreads();

    // A fragment 16x32 (contiguous k order: k = 16*g + e, same as B)
    v16bf a;
    const int mrow = wave * 16 + l15;
#pragma unroll
    for (int e = 0; e < 16; ++e) {
      int j = e >> 1, hh = e & 1;
      int k = g * 16 + 2 * j + hh;
      a[e] = lA[mrow][k];
    }
    // B fragments (32x16: lanes 0-15 K=0..15, lanes 16-31 K=16..31)
    v16bf bfrag[4];
#pragma unroll
    for (int ni = 0; ni < 4; ++ni) {
      const int nrow = ni * 16 + l15;
#pragma unroll
      for (int e = 0; e < 16; ++e) bfrag[ni][e] = lW[nrow][g * 16 + e];
    }
#pragma unroll
    for (int ni = 0; ni < 4; ++ni)
      acc[ni] = __builtin_amdgcn_wmma_f32_16x16x32_bf16(
          false, a, false, bfrag[ni], (short)0, acc[ni], false, false);
    __syncthreads();
  }

  // C/D layout: VGPR v -> M = v + 8*(lane>=16), N = lane%16
  const int rbase = g * 8;
#pragma unroll
  for (int ni = 0; ni < 4; ++ni) {
    int n = n0 + ni * 16 + l15;
    if (n < Nreal) {
#pragma unroll
      for (int v = 0; v < 8; ++v) {
        int m = m0 + wave * 16 + rbase + v;
        float r = acc[ni][v];
        if (addend) r += addend[(size_t)m * Npad + n];
        if (bias) r += bias[n];
        if (act) r = fmaxf(r, 0.f);
        Y[(size_t)m * Npad + n] = r;
        if (Ybf) Ybf[(size_t)m * Npad + n] = (bf16_t)r;
      }
    }
  }
}

__global__ __launch_bounds__(256) void k_gru(const float* __restrict__ gates,
                                             float* __restrict__ h,
                                             bf16_t* __restrict__ xs_bf,
                                             bf16_t* __restrict__ hrelu_bf) {
  size_t i = (size_t)blockIdx.x * blockDim.x + threadIdx.x;
  if (i >= (size_t)BATCH * HID) return;
  int b = (int)(i / HID), j = (int)(i % HID);
  const float* gb = gates + (size_t)b * NGATE;
  float r = 1.f / (1.f + __expf(-gb[j]));
  float z = 1.f / (1.f + __expf(-gb[512 + j]));
  float n = tanhf(gb[1024 + j] + r * gb[1536 + j]);
  float hn = (1.f - z) * n + z * h[i];
  h[i] = hn;
  xs_bf[(size_t)b * KGATE + j] = (bf16_t)hn;
  hrelu_bf[i] = (bf16_t)fmaxf(hn, 0.f);
}

__global__ __launch_bounds__(256) void k_out(int t, const float* __restrict__ o2,
                                             float* __restrict__ o_prev,
                                             float* __restrict__ out) {
  size_t i = (size_t)blockIdx.x * blockDim.x + threadIdx.x;
  if (i >= (size_t)BATCH * OUT_SZ) return;
  int b = (int)(i / OUT_SZ), j = (int)(i % OUT_SZ);
  float v = o2[(size_t)b * 64 + j];
  o_prev[i] = v;
  if (j < 40) {
    out[((size_t)b * NLINES + t) * 40 + j] = v;
  } else if (j == 40) {
    float a = v, c = o2[(size_t)b * 64 + 41];
    float m = fmaxf(a, c);
    float e0 = __expf(a - m), e1 = __expf(c - m);
    float s = e0 + e1;
    size_t base = (size_t)BATCH * NLINES * 40 + ((size_t)b * NLINES + t) * 2;
    out[base] = e0 / s;
    out[base + 1] = e1 / s;
  }
}

extern "C" void kernel_launch(void* const* d_in, const int* in_sizes, int n_in,
                              void* d_out, int out_size, void* d_ws, size_t ws_size,
                              hipStream_t stream) {
  const float* img       = (const float*)d_in[0];
  const float* gt        = (const float*)d_in[1];
  const int*   line_prob = (const int*)d_in[2];
  const float* samp      = (const float*)d_in[3];
  const float* att_w     = (const float*)d_in[4];
  const float* w_ih      = (const float*)d_in[6];
  const float* w_hh      = (const float*)d_in[7];
  const float* b_ih      = (const float*)d_in[8];
  const float* b_hh      = (const float*)d_in[9];
  const float* l1w       = (const float*)d_in[10];
  const float* l1b       = (const float*)d_in[11];
  const float* l2w       = (const float*)d_in[12];
  const float* l2b       = (const float*)d_in[13];
  float* out = (float*)d_out;
  (void)in_sizes; (void)n_in; (void)out_size; (void)ws_size;

  char* ws = (char*)d_ws;
  size_t off = 0;
  auto carve = [&](size_t bytes) -> char* {
    off = (off + 255) & ~(size_t)255;
    char* p = ws + off;
    off += bytes;
    return p;
  };
  bf16_t* att_feat_bf = (bf16_t*)carve((size_t)BATCH * FEAT * 2);
  bf16_t* Wfeat       = (bf16_t*)carve((size_t)NGATE * FEAT * 2);
  float*  bias_big    = (float*)carve((size_t)NGATE * 4);
  float*  gi_base     = (float*)carve((size_t)BATCH * NGATE * 4);
  bf16_t* Wgates      = (bf16_t*)carve((size_t)NGATE * KGATE * 2);
  bf16_t* lin1bf      = (bf16_t*)carve((size_t)512 * 512 * 2);
  bf16_t* lin2bf      = (bf16_t*)carve((size_t)64 * 512 * 2);
  float*  h           = (float*)carve((size_t)BATCH * HID * 4);
  bf16_t* xs_bf       = (bf16_t*)carve((size_t)BATCH * KGATE * 2);
  float*  gates       = (float*)carve((size_t)BATCH * NGATE * 4);
  float*  o1          = (float*)carve((size_t)BATCH * 512 * 4);
  bf16_t* hrelu_bf    = (bf16_t*)carve((size_t)BATCH * 512 * 2);
  bf16_t* o1relu_bf   = (bf16_t*)carve((size_t)BATCH * 512 * 2);
  float*  o2          = (float*)carve((size_t)BATCH * 64 * 4);
  float*  o_prev      = (float*)carve((size_t)BATCH * OUT_SZ * 4);

  k_attfeat<<<BATCH, 256, 0, stream>>>(img, att_w, att_feat_bf);
  {
    size_t total = (size_t)NGATE * FEAT + NGATE + (size_t)NGATE * KGATE +
                   (size_t)512 * 512 + (size_t)64 * 512;
    k_prep_w<<<(int)((total + 255) / 256), 256, 0, stream>>>(
        w_ih, w_hh, b_ih, b_hh, l1w, l2w, Wfeat, bias_big, Wgates, lin1bf, lin2bf);
  }
  {
    size_t total = (size_t)BATCH * (HID + OUT_SZ + KGATE);
    k_init<<<(int)((total + 255) / 256), 256, 0, stream>>>(h, o_prev, xs_bf);
  }
  // gi_base = att_feat @ Wfeat^T + bias_big   (B x 2048, K = 512)
  k_gemm<<<dim3(BATCH / 128, NGATE / 64), 256, 0, stream>>>(
      att_feat_bf, Wfeat, nullptr, bias_big, gi_base, nullptr, FEAT, NGATE, NGATE, 0);

  const int nSS = (BATCH * OUT_SZ + 255) / 256;
  const int nGR = (BATCH * HID + 255) / 256;
  for (int t = 0; t < NLINES; ++t) {
    k_ss<<<nSS, 256, 0, stream>>>(t, samp, gt, line_prob, o_prev, xs_bf);
    // gates = [h | ss] @ Wgates^T + gi_base   (B x 2048, K = 576)
    k_gemm<<<dim3(BATCH / 128, NGATE / 64), 256, 0, stream>>>(
        xs_bf, Wgates, gi_base, nullptr, gates, nullptr, KGATE, NGATE, NGATE, 0);
    k_gru<<<nGR, 256, 0, stream>>>(gates, h, xs_bf, hrelu_bf);
    // lin1: relu(h) @ l1w^T + l1b, ReLU -> bf16
    k_gemm<<<dim3(BATCH / 128, 512 / 64), 256, 0, stream>>>(
        hrelu_bf, lin1bf, nullptr, l1b, o1, o1relu_bf, 512, 512, 512, 1);
    // lin2: relu(o1) @ l2w^T + l2b  (Npad 64, Nreal 42)
    k_gemm<<<dim3(BATCH / 128, 1), 256, 0, stream>>>(
        o1relu_bf, lin2bf, nullptr, l2b, o2, nullptr, 512, 64, OUT_SZ, 0);
    k_out<<<nSS, 256, 0, stream>>>(t, o2, o_prev, out);
  }
}
